// SelectiveSSM_30966714204478
// MI455X (gfx1250) — matrix operand order repaired and checked by reading, not compiled, not run on hardware
//
#include <hip/hip_runtime.h>
#include <hip/hip_bf16.h>
#include <math.h>
#include <stdint.h>

typedef __attribute__((ext_vector_type(16))) __bf16 v16bf;
typedef __attribute__((ext_vector_type(8)))  float  v8f;
typedef int v4i __attribute__((__vector_size__(16)));
typedef unsigned long long ull;

#define BM 128
#define BN 128
#define BKK 32
#define LDS_STRIDE 40            // halves/row: 32 + 8 pad -> 80B rows, 16B-aligned chunks,
                                 // 20-dword stride = conflict-free fragment reads
#define LDSU (BM * LDS_STRIDE / 4)   // ulongs per tile buffer (1280)

#if __has_builtin(__builtin_amdgcn_global_load_async_to_lds_b128) && \
    __has_builtin(__builtin_amdgcn_s_wait_asynccnt)
#define HAVE_ASYNC 1
#else
#define HAVE_ASYNC 0
#endif

// prototype (from clang diagnostic): (v4i AS1*, v4i AS3*, int offset, int cpol)
#define GPTR(p) ((__attribute__((address_space(1))) v4i*)(uintptr_t)(p))
#define LPTR(p) ((__attribute__((address_space(3))) v4i*)(uintptr_t)(p))

union Frag {
    ull   u[4];
    v16bf v;
};

__device__ __forceinline__ float sigmf(float x) { return 1.0f / (1.0f + __expf(-x)); }

// ---------------------------------------------------------------------------
// Generic bf16 GEMM:  C(M x N) = A(M x K, row-major bf16) * W(N x K, row-major bf16)^T
// Double-buffered LDS, async global->LDS staging when available.
// MODE 0: split epilogue -> o0 = bf16[M x N/2] (cols < N/2), o1 = bf16[M x N/2]
// MODE 1: o0 = f32[M x N], col-guarded (N not multiple of 128)
// MODE 2: o0 = f32[M x N]
// ---------------------------------------------------------------------------
template <int MODE>
__global__ __launch_bounds__(256) void gemm_bf16_wmma(
    const __hip_bfloat16* __restrict__ A,
    const __hip_bfloat16* __restrict__ W,
    void* __restrict__ o0, void* __restrict__ o1,
    int M, int N, int K)
{
    __shared__ ull lA[2 * LDSU];   // 2 x 10240 B
    __shared__ ull lB[2 * LDSU];

    const int tid   = threadIdx.x;
    const int lane  = tid & 31;
    const int wave  = tid >> 5;
    const int waveM = wave & 1;    // 2 waves along M
    const int waveN = wave >> 1;   // 4 waves along N
    const int blockM = blockIdx.y * BM;
    const int blockN = blockIdx.x * BN;

    const int srow = tid >> 1;     // 0..127 : staged row
    const int sh   = tid & 1;      // which 16-half (32B) segment of the row

    // Per-thread staging source/dest. B rows are clamped: garbage only reaches
    // output columns >= N, which the epilogue never stores (WMMA is col-separable).
    int nclamp = blockN + srow; if (nclamp >= N) nclamp = N - 1;
    const char* gAb = (const char*)(A + (size_t)(blockM + srow) * K + sh * 16);
    const char* gBb = (const char*)(W + (size_t)nclamp * K + sh * 16);

    auto stage = [&](int k0, int buf) {
        const char* ga = gAb + (size_t)k0 * 2;
        const char* gb = gBb + (size_t)k0 * 2;
        ull* da = lA + buf * LDSU + srow * 10 + sh * 4;
        ull* db = lB + buf * LDSU + srow * 10 + sh * 4;
#if HAVE_ASYNC
        __builtin_amdgcn_global_load_async_to_lds_b128(GPTR(ga),      LPTR(da),     0, 0);
        __builtin_amdgcn_global_load_async_to_lds_b128(GPTR(ga + 16), LPTR(da + 2), 0, 0);
        __builtin_amdgcn_global_load_async_to_lds_b128(GPTR(gb),      LPTR(db),     0, 0);
        __builtin_amdgcn_global_load_async_to_lds_b128(GPTR(gb + 16), LPTR(db + 2), 0, 0);
#else
        const ull* a8 = (const ull*)ga;
        const ull* b8 = (const ull*)gb;
        da[0] = a8[0]; da[1] = a8[1]; da[2] = a8[2]; da[3] = a8[3];
        db[0] = b8[0]; db[1] = b8[1]; db[2] = b8[2]; db[3] = b8[3];
#endif
        if (k0 + 2 * BKK < K) {
            __builtin_prefetch(ga + 128, 0, 0);   // warm L2 two tiles ahead
            __builtin_prefetch(gb + 128, 0, 0);
        }
    };

    v8f acc[4][2];
#pragma unroll
    for (int i = 0; i < 4; ++i)
#pragma unroll
        for (int j = 0; j < 2; ++j)
            acc[i][j] = (v8f){0.f, 0.f, 0.f, 0.f, 0.f, 0.f, 0.f, 0.f};

    // prologue: stage first K-tile into buffer 0
    stage(0, 0);
#if HAVE_ASYNC
    __builtin_amdgcn_s_wait_asynccnt(0);
#endif
    __syncthreads();

    const int hi = lane >> 4;
    const int r  = lane & 15;

    int buf = 0;
    for (int k0 = 0; k0 < K; k0 += BKK) {
        // ---- load fragments from current buffer (CDNA5 wave32 WMMA layouts) ----
        const ull* cA = lA + buf * LDSU;
        const ull* cB = lB + buf * LDSU;

        Frag a[4];
#pragma unroll
        for (int i = 0; i < 4; ++i) {
            const int row = waveM * 64 + i * 16 + r;
            const int idx = row * 10 + hi * 2;           // halves: row*40 + hi*8
            a[i].u[0] = cA[idx];
            a[i].u[1] = cA[idx + 1];
            a[i].u[2] = cA[idx + 4];                     // +16 halves
            a[i].u[3] = cA[idx + 5];
        }
        Frag b[2];
#pragma unroll
        for (int j = 0; j < 2; ++j) {
            const int col = waveN * 32 + j * 16 + r;
            const int idx = col * 10 + hi * 2;           // halves: col*40 + hi*8
            b[j].u[0] = cB[idx];
            b[j].u[1] = cB[idx + 1];
            b[j].u[2] = cB[idx + 4];
            b[j].u[3] = cB[idx + 5];
        }

        // ---- kick off next tile's staging into the other buffer (safe: its
        // previous readers all passed last iteration's barrier) ----
        const bool more = (k0 + BKK) < K;
        if (more) stage(k0 + BKK, buf ^ 1);

        // ---- matrix math overlaps the in-flight async loads ----
#pragma unroll
        for (int i = 0; i < 4; ++i)
#pragma unroll
            for (int j = 0; j < 2; ++j)
                acc[i][j] = __builtin_amdgcn_wmma_f32_16x16x32_bf16(
                    false, a[i].v, false, b[j].v, (short)0, acc[i][j], false, false);

#if HAVE_ASYNC
        if (more) __builtin_amdgcn_s_wait_asynccnt(0);
#endif
        __syncthreads();   // next buffer visible to everyone; frag reads of cur done
        buf ^= 1;
    }

    // ---- epilogue: C/D layout -> VGPR v = row v (lanes<16) / v+8 (lanes>=16) ----
    const int rr = hi * 8;
    const int cc = r;
#pragma unroll
    for (int i = 0; i < 4; ++i) {
#pragma unroll
        for (int j = 0; j < 2; ++j) {
            const int gr0 = blockM + waveM * 64 + i * 16 + rr;
            const int gc  = blockN + waveN * 32 + j * 16 + cc;
#pragma unroll
            for (int v = 0; v < 8; ++v) {
                const int   grow = gr0 + v;
                const float val  = acc[i][j][v];
                if (MODE == 0) {
                    const int half = N >> 1;
                    if (gc < half)
                        ((__hip_bfloat16*)o0)[(size_t)grow * half + gc] = __float2bfloat16(val);
                    else
                        ((__hip_bfloat16*)o1)[(size_t)grow * half + (gc - half)] = __float2bfloat16(val);
                } else if (MODE == 1) {
                    if (gc < N)
                        ((float*)o0)[(size_t)grow * N + gc] = val;
                } else {
                    ((float*)o0)[(size_t)grow * N + gc] = val;
                }
            }
        }
    }
}

// ---------------------------------------------------------------------------
// f32 -> bf16 conversion
// ---------------------------------------------------------------------------
__global__ void cvt_f32_bf16(const float* __restrict__ in, __hip_bfloat16* __restrict__ out, int n)
{
    int i = blockIdx.x * blockDim.x + threadIdx.x;
    if (i < n) out[i] = __float2bfloat16(in[i]);
}

// ---------------------------------------------------------------------------
// Depthwise causal conv (d_conv=4) + bias + SiLU, bf16 in / bf16 out
// ---------------------------------------------------------------------------
__global__ void conv_silu_kernel(const __hip_bfloat16* __restrict__ xb,
                                 const float* __restrict__ cw,
                                 const float* __restrict__ cb,
                                 __hip_bfloat16* __restrict__ xc,
                                 int Bsz, int L, int C)
{
    int idx = blockIdx.x * blockDim.x + threadIdx.x;
    int total = Bsz * L * C;
    if (idx >= total) return;
    int c = idx % C;
    int l = (idx / C) % L;
    int b = idx / (C * L);

    float acc = cb[c];
#pragma unroll
    for (int k = 0; k < 4; ++k) {
        int ls = l - 3 + k;
        if (ls >= 0)
            acc += cw[c * 4 + k] * __bfloat162float(xb[((size_t)b * L + ls) * C + c]);
    }
    acc = acc * sigmf(acc);          // SiLU
    xc[idx] = __float2bfloat16(acc);
}

// ---------------------------------------------------------------------------
// SSM pointwise: bc = <Bm,Cm>; y = (softplus(delta)*bc + D)*xc * silu(z)
// One block per token.
// ---------------------------------------------------------------------------
__global__ void ssm_pointwise_kernel(const float* __restrict__ proj,        // [tokens x 2080]
                                     const __hip_bfloat16* __restrict__ xc, // [tokens x C]
                                     const __hip_bfloat16* __restrict__ z,  // [tokens x C]
                                     const float* __restrict__ Dp,          // [C]
                                     __hip_bfloat16* __restrict__ y,        // [tokens x C]
                                     int C, int Nproj)
{
    const int t = blockIdx.x;
    const float* pr = proj + (size_t)t * Nproj;

    float bc = 0.f;
#pragma unroll
    for (int s = 0; s < 16; ++s) bc += pr[s] * pr[16 + s];

    for (int c = threadIdx.x; c < C; c += blockDim.x) {
        float d = pr[32 + c];
        float delta = (d > 20.f) ? d : log1pf(__expf(d));
        float xcf = __bfloat162float(xc[(size_t)t * C + c]);
        float zf  = __bfloat162float(z[(size_t)t * C + c]);
        float yv  = (delta * bc + Dp[c]) * xcf;
        yv *= zf * sigmf(zf);        // * SiLU(z)
        y[(size_t)t * C + c] = __float2bfloat16(yv);
    }
}

// ---------------------------------------------------------------------------
extern "C" void kernel_launch(void* const* d_in, const int* in_sizes, int n_in,
                              void* d_out, int out_size, void* d_ws, size_t ws_size,
                              hipStream_t stream)
{
    (void)in_sizes; (void)n_in; (void)out_size; (void)ws_size;

    const float* x      = (const float*)d_in[0];   // [2,4096,1024]
    const float* W_in   = (const float*)d_in[1];   // [4096,1024]
    const float* conv_w = (const float*)d_in[2];   // [2048,4]
    const float* conv_b = (const float*)d_in[3];   // [2048]
    const float* W_xprj = (const float*)d_in[4];   // [2080,2048]
    const float* Dp     = (const float*)d_in[5];   // [2048]
    const float* W_out  = (const float*)d_in[6];   // [1024,2048]

    constexpr int Bsz = 2, L = 4096, dm = 1024, di = 2048, dstate = 16;
    constexpr int M  = Bsz * L;          // 8192 tokens
    constexpr int E1 = 2 * di;           // 4096
    constexpr int E2 = 2 * dstate + di;  // 2080

    char* ws = (char*)d_ws;
    size_t off = 0;
    auto alloc = [&](size_t bytes) -> void* {
        void* p = ws + off;
        off = (off + bytes + 255) & ~(size_t)255;
        return p;
    };

    __hip_bfloat16* x_bf   = (__hip_bfloat16*)alloc((size_t)M * dm * 2);
    __hip_bfloat16* win_bf = (__hip_bfloat16*)alloc((size_t)E1 * dm * 2);
    __hip_bfloat16* wxp_bf = (__hip_bfloat16*)alloc((size_t)E2 * di * 2);
    __hip_bfloat16* wou_bf = (__hip_bfloat16*)alloc((size_t)dm * di * 2);
    __hip_bfloat16* xb_bf  = (__hip_bfloat16*)alloc((size_t)M * di * 2);
    __hip_bfloat16* z_bf   = (__hip_bfloat16*)alloc((size_t)M * di * 2);
    __hip_bfloat16* xc_bf  = (__hip_bfloat16*)alloc((size_t)M * di * 2);
    float*          proj   = (float*)alloc((size_t)M * E2 * 4);
    __hip_bfloat16* y_bf   = (__hip_bfloat16*)alloc((size_t)M * di * 2);

    // 1) convert inputs to bf16
    {
        int n;
        n = M * dm;   cvt_f32_bf16<<<(n + 255) / 256, 256, 0, stream>>>(x,      x_bf,   n);
        n = E1 * dm;  cvt_f32_bf16<<<(n + 255) / 256, 256, 0, stream>>>(W_in,   win_bf, n);
        n = E2 * di;  cvt_f32_bf16<<<(n + 255) / 256, 256, 0, stream>>>(W_xprj, wxp_bf, n);
        n = dm * di;  cvt_f32_bf16<<<(n + 255) / 256, 256, 0, stream>>>(W_out,  wou_bf, n);
    }

    // 2) GEMM1: xz = x * W_in^T  -> split into xb, z (bf16)
    gemm_bf16_wmma<0><<<dim3(E1 / BN, M / BM), 256, 0, stream>>>(
        x_bf, win_bf, xb_bf, z_bf, M, E1, dm);

    // 3) depthwise conv + SiLU -> xc
    {
        int total = M * di;
        conv_silu_kernel<<<(total + 255) / 256, 256, 0, stream>>>(
            xb_bf, conv_w, conv_b, xc_bf, Bsz, L, di);
    }

    // 4) GEMM2: proj = xc * W_xproj^T  (f32, N=2080 guarded)
    gemm_bf16_wmma<1><<<dim3((E2 + BN - 1) / BN, M / BM), 256, 0, stream>>>(
        xc_bf, wxp_bf, proj, nullptr, M, E2, di);

    // 5) SSM pointwise -> y (bf16)
    ssm_pointwise_kernel<<<M, 256, 0, stream>>>(proj, xc_bf, z_bf, Dp, y_bf, di, E2);

    // 6) GEMM3: out = y * W_out^T  (f32 to d_out)
    gemm_bf16_wmma<2><<<dim3(dm / BN, M / BM), 256, 0, stream>>>(
        y_bf, wou_bf, d_out, nullptr, M, dm, di);
}
